// ImageTriplaneGenerator_42666205118755
// MI455X (gfx1250) — matrix operand order repaired and checked by reading, not compiled, not run on hardware
//
#include <hip/hip_runtime.h>
#include <hip/hip_bf16.h>
#include <math.h>

// ---------------- problem constants ----------------
#define NLAYER   3
#define CDIM     196
#define C4DIM    784
#define NHEAD    4
#define DHEAD    49
#define LTOK     49152          // B * L
#define NV_SLOTS 4
#define NS_SLOTS 16
#define QW_LD    800            // 784 qw cols + 4 qb cols + pad to x16
#define QN_LD    208            // qn row stride (16B-aligned rows for async b128)
#define KP196    208            // padded K stride for K=196 weight rows
#define NTILE    13             // 13*16 = 208 >= 196 (full-N coverage)
#define ALD      48             // LDS row stride (bf16) for A tiles (96B, 32B-aligned)
#define BLD      48             // LDS row stride (bf16) for B tiles

typedef __bf16 bf16_t;
typedef __attribute__((ext_vector_type(16))) __bf16 v16bf;
typedef __attribute__((ext_vector_type(8)))  float  v8f;

__device__ __forceinline__ bf16_t f2bf(float f) {
  unsigned u = __builtin_bit_cast(unsigned, f);
  unsigned r = (u + 0x7FFFu + ((u >> 16) & 1u)) >> 16;   // RNE
  unsigned short s = (unsigned short)r;
  return __builtin_bit_cast(bf16_t, s);
}
__device__ __forceinline__ float bf2f(bf16_t b) {
  unsigned short s = __builtin_bit_cast(unsigned short, b);
  unsigned u = ((unsigned)s) << 16;
  return __builtin_bit_cast(float, u);
}
__device__ __forceinline__ unsigned pk2(float a, float b) {
  unsigned lo = __builtin_bit_cast(unsigned short, f2bf(a));
  unsigned hi = __builtin_bit_cast(unsigned short, f2bf(b));
  return lo | (hi << 16);
}
__device__ __forceinline__ v8f wmma_bf16(v16bf a, v16bf b, v8f c) {
  return __builtin_amdgcn_wmma_f32_16x16x32_bf16(false, a, false, b,
                                                 (short)0, c, false, false);
}
// swap the two middle 8-element K blocks (self-inverse) so each lane's
// A fragment (ISA 7.12.2 16-bit A 16x32 layout) is one contiguous 32B chunk.
__device__ __forceinline__ int permpos(int x) {
  int b = x & 24;
  return (b == 8) ? x + 8 : (b == 16) ? x - 8 : x;
}

// ---- CDNA5 async global->LDS copy (ASYNCcnt), 16B per lane ----
__device__ __forceinline__ void async_copy_b128(bf16_t* lds_dst, const bf16_t* gsrc) {
  unsigned lds = (unsigned)(uintptr_t)lds_dst;   // low 32 bits = LDS byte offset
  asm volatile("global_load_async_to_lds_b128 %0, %1, off"
               :: "v"(lds), "v"(gsrc) : "memory");
}
__device__ __forceinline__ void wait_async_lds() {
  asm volatile("s_wait_asynccnt 0" ::: "memory");
}

// ---- stage one 128x32 A tile into permuted LDS layout ----
// bf16 path: pure 16B async copies (rows must be 16B-aligned -> padded lda).
template<int A_BF16>
__device__ __forceinline__ void stage_A(bf16_t* As, const void* Ap, int m0blk,
                                        int k0, int M, int K, int lda, int tid) {
  int r = tid >> 1, cb = (tid & 1) * 16;
  int gm = m0blk + r;
  if (gm < M && k0 + 32 <= K) {
    if (A_BF16) {
      const bf16_t* src = (const bf16_t*)Ap + (size_t)gm * lda + k0 + cb;
      async_copy_b128(&As[r * ALD + permpos(cb)],     src);
      async_copy_b128(&As[r * ALD + permpos(cb + 8)], src + 8);
    } else {
      const float* src = (const float*)Ap + (size_t)gm * lda + k0 + cb;
      unsigned pk[8];
      #pragma unroll
      for (int g = 0; g < 4; ++g) {
        float4 f = *(const float4*)(src + 4 * g);
        pk[2 * g] = pk2(f.x, f.y); pk[2 * g + 1] = pk2(f.z, f.w);
      }
      uint4 q0; q0.x = pk[0]; q0.y = pk[1]; q0.z = pk[2]; q0.w = pk[3];
      uint4 q1; q1.x = pk[4]; q1.y = pk[5]; q1.z = pk[6]; q1.w = pk[7];
      *(uint4*)&As[r * ALD + permpos(cb)]     = q0;
      *(uint4*)&As[r * ALD + permpos(cb + 8)] = q1;
    }
  } else {
    #pragma unroll
    for (int i = 0; i < 16; ++i) {
      int k = k0 + cb + i;
      float v = 0.f;
      if (gm < M && k < K) {
        if (A_BF16) v = bf2f(((const bf16_t*)Ap)[(size_t)gm * lda + k]);
        else        v = ((const float*)Ap)[(size_t)gm * lda + k];
      }
      As[r * ALD + permpos(cb + i)] = f2bf(v);
    }
  }
}

// ---- stage one transposed B chunk: Bt[n][kb..kb+7] = WT[gn][gk..gk+7] ----
__device__ __forceinline__ void stage_B_slot(bf16_t* Bt, const bf16_t* WT,
                                             int n, int kb, int n0blk, int k0,
                                             int N, int K, int ldw) {
  int gn = n0blk + n, gk = k0 + kb;
  bf16_t* dst = &Bt[n * BLD + permpos(kb)];
  if (gn < N && gk + 8 <= K) {
    async_copy_b128(dst, WT + (size_t)gn * ldw + gk);    // 16B async copy
  } else {
    #pragma unroll
    for (int i = 0; i < 8; ++i) {
      int k = gk + i;
      dst[i] = (gn < N && k < K) ? WT[(size_t)gn * ldw + k] : f2bf(0.f);
    }
  }
}

// ---------------- generic tiled GEMM: out(bf16) = epi(A @ W + bias) ----------------
// BM=128 (8 waves x 16 rows), BN=64, K step 32, double-buffered async staging.
template<int A_BF16, int DO_GELU>
__global__ __launch_bounds__(256)
void gemm_tile_kernel(const void* __restrict__ Ap, const bf16_t* __restrict__ WT,
                      const float* __restrict__ bias, bf16_t* __restrict__ Out,
                      int M, int N, int K, int lda, int ldw, int ldo) {
  __shared__ bf16_t As[2][128 * ALD];
  __shared__ bf16_t Bt[2][64 * BLD];
  const int tid  = threadIdx.x;
  const int wave = tid >> 5, lane = tid & 31;
  const int half = lane >> 4, lm = lane & 15;
  const int m0blk = blockIdx.x * 128, n0blk = blockIdx.y * 64;
  const int mw = wave * 16;
  const int nsteps = (K + 31) / 32;

  v8f acc[4] = {};
  stage_A<A_BF16>(As[0], Ap, m0blk, 0, M, K, lda, tid);
  stage_B_slot(Bt[0], WT, tid & 63, (tid >> 6) * 8, n0blk, 0, N, K, ldw);
  wait_async_lds();
  __syncthreads();
  int cur = 0;
  for (int s = 0; s < nsteps; ++s) {
    if (s + 1 < nsteps) {    // prefetch next tiles (async, overlaps WMMA below)
      stage_A<A_BF16>(As[cur ^ 1], Ap, m0blk, (s + 1) * 32, M, K, lda, tid);
      stage_B_slot(Bt[cur ^ 1], WT, tid & 63, (tid >> 6) * 8, n0blk, (s + 1) * 32, N, K, ldw);
    }
    v16bf af = *(const v16bf*)&As[cur][(mw + lm) * ALD + 16 * half];
    #pragma unroll
    for (int t = 0; t < 4; ++t) {
      v16bf bv = *(const v16bf*)&Bt[cur][(t * 16 + lm) * BLD + 16 * half];
      acc[t] = wmma_bf16(af, bv, acc[t]);
    }
    wait_async_lds();
    __syncthreads();
    cur ^= 1;
  }
  #pragma unroll
  for (int t = 0; t < 4; ++t) {
    int n = n0blk + t * 16 + lm;
    float bv = (bias != nullptr && n < N) ? bias[n] : 0.f;
    #pragma unroll
    for (int r = 0; r < 8; ++r) {
      int m = m0blk + mw + r + 8 * half;   // C/D layout: lanes16-31 hold M=r+8
      if (m < M && n < N) {
        float x = acc[t][r] + bv;
        if (DO_GELU) x = 0.5f * x * (1.f + erff(x * 0.70710678f));
        Out[(size_t)m * ldo + n] = f2bf(x);
      }
    }
  }
}

// -------- full-row GEMM + residual + LayerNorm epilogue: q = LN(res + A@W + b) --------
__global__ __launch_bounds__(256)
void gemm_rowln_kernel(const bf16_t* __restrict__ A, const bf16_t* __restrict__ WT,
                       const float* __restrict__ bias, const float* __restrict__ res,
                       const float* __restrict__ gamma, const float* __restrict__ beta,
                       float* __restrict__ Out, int M, int K) {
  const int N = CDIM;
  __shared__ bf16_t As[2][128 * ALD];     // 24.0 KB
  __shared__ bf16_t Bt[2][208 * BLD];     // 39.0 KB  (total 63 KB)
  const int tid  = threadIdx.x;
  const int wave = tid >> 5, lane = tid & 31;
  const int half = lane >> 4, lm = lane & 15;
  const int m0blk = blockIdx.x * 128;
  const int mw = wave * 16;
  const int nsteps = (K + 31) / 32;

  v8f acc[NTILE] = {};
  stage_A<1>(As[0], A, m0blk, 0, M, K, K, tid);
  #pragma unroll
  for (int slot = tid; slot < 832; slot += 256)
    stage_B_slot(Bt[0], WT, slot % 208, (slot / 208) * 8, 0, 0, N, K, K);
  wait_async_lds();
  __syncthreads();
  int cur = 0;
  for (int s = 0; s < nsteps; ++s) {
    if (s + 1 < nsteps) {
      int k0n = (s + 1) * 32;
      stage_A<1>(As[cur ^ 1], A, m0blk, k0n, M, K, K, tid);
      #pragma unroll
      for (int slot = tid; slot < 832; slot += 256)
        stage_B_slot(Bt[cur ^ 1], WT, slot % 208, (slot / 208) * 8, 0, k0n, N, K, K);
    }
    v16bf af = *(const v16bf*)&As[cur][(mw + lm) * ALD + 16 * half];
    #pragma unroll
    for (int t = 0; t < NTILE; ++t) {
      v16bf bv = *(const v16bf*)&Bt[cur][(t * 16 + lm) * BLD + 16 * half];
      acc[t] = wmma_bf16(af, bv, acc[t]);
    }
    wait_async_lds();
    __syncthreads();
    cur ^= 1;
  }
  // bias + residual (padded cols stay exactly 0 so they don't disturb row stats)
  #pragma unroll
  for (int t = 0; t < NTILE; ++t) {
    int n = t * 16 + lm;
    float bv = (n < N) ? bias[n] : 0.f;
    #pragma unroll
    for (int r = 0; r < 8; ++r) {
      int m = m0blk + mw + r + 8 * half;
      float v = acc[t][r] + bv;
      if (n < N && m < M) v += res[(size_t)m * N + n]; else v = 0.f;
      acc[t][r] = v;
    }
  }
  // per-row LayerNorm: reduce across the 16 lanes of each half (xor 1,2,4,8)
  #pragma unroll
  for (int r = 0; r < 8; ++r) {
    float s = 0.f, s2 = 0.f;
    #pragma unroll
    for (int t = 0; t < NTILE; ++t) { float v = acc[t][r]; s += v; s2 += v * v; }
    #pragma unroll
    for (int msk = 8; msk >= 1; msk >>= 1) {
      s  += __shfl_xor(s,  msk, 32);
      s2 += __shfl_xor(s2, msk, 32);
    }
    float mean = s * (1.f / N);
    float var  = s2 * (1.f / N) - mean * mean;
    float rstd = rsqrtf(var + 1e-5f);
    int m = m0blk + mw + r + 8 * half;
    #pragma unroll
    for (int t = 0; t < NTILE; ++t) {
      int n = t * 16 + lm;
      if (n < N && m < M)
        Out[(size_t)m * N + n] = (acc[t][r] - mean) * rstd * gamma[n] + beta[n];
    }
  }
}

// ---------------- qn = LN(q)*g + b  (bf16 out, ld=QN_LD), one wave per token ----------------
__global__ __launch_bounds__(256)
void ln_act_kernel(const float* __restrict__ q, const float* __restrict__ g,
                   const float* __restrict__ b, bf16_t* __restrict__ outbf, int M) {
  int wave = blockIdx.x * 8 + (threadIdx.x >> 5);
  int lane = threadIdx.x & 31;
  if (wave >= M) return;
  const float* row = q + (size_t)wave * CDIM;
  float v[7]; float s = 0.f, s2 = 0.f;
  #pragma unroll
  for (int j = 0; j < 7; ++j) {
    int e = j * 32 + lane;
    float x = (e < CDIM) ? row[e] : 0.f;
    v[j] = x; s += x; s2 += x * x;
  }
  #pragma unroll
  for (int msk = 16; msk >= 1; msk >>= 1) {
    s += __shfl_xor(s, msk, 32); s2 += __shfl_xor(s2, msk, 32);
  }
  float mean = s * (1.f / CDIM);
  float var  = s2 * (1.f / CDIM) - mean * mean;
  float rstd = rsqrtf(var + 1e-5f);
  #pragma unroll
  for (int j = 0; j < 7; ++j) {
    int e = j * 32 + lane;
    if (e < CDIM)
      outbf[(size_t)wave * QN_LD + e] = f2bf((v[j] - mean) * rstd * g[e] + b[e]);
  }
  if (lane < QN_LD - CDIM)               // zero row padding
    outbf[(size_t)wave * QN_LD + CDIM + lane] = f2bf(0.f);
}

// ---- WqT [800][208] bf16: row j=h*196+e, col c -> Wk[e,c] iff head(c)==h; rows 784+h -> bk ----
__global__ void make_wqT_kernel(const float* __restrict__ Wk, const float* __restrict__ bk,
                                bf16_t* __restrict__ WqT) {
  int idx = blockIdx.x * blockDim.x + threadIdx.x;
  if (idx >= QW_LD * KP196) return;
  int j = idx / KP196, c = idx % KP196;
  float v = 0.f;
  if (c < CDIM) {
    int hc = c / DHEAD;
    if (j < C4DIM)          { int h = j / CDIM, e = j % CDIM; if (h == hc) v = Wk[(size_t)e * CDIM + c]; }
    else if (j < C4DIM + 4) { int h = j - C4DIM;              if (h == hc) v = bk[c]; }
  }
  WqT[idx] = f2bf(v);
}

// ---- WvT [196][784] bf16: row c, col j=h*196+e -> Wv[e,c] iff head(c)==h ----
__global__ void make_wvT_kernel(const float* __restrict__ Wv, bf16_t* __restrict__ WvT) {
  int idx = blockIdx.x * blockDim.x + threadIdx.x;
  if (idx >= CDIM * C4DIM) return;
  int c = idx / C4DIM, j = idx % C4DIM;
  int h = j / CDIM, e = j % CDIM;
  WvT[idx] = f2bf((h == c / DHEAD) ? Wv[(size_t)e * CDIM + c] : 0.f);
}

// ---- generic transpose+convert with K padding: WT[n][k<K] = W[k][n], else 0 ----
__global__ void transpose_bf16_kernel(const float* __restrict__ W, bf16_t* __restrict__ WT,
                                      int K, int N, int ldk) {
  int idx = blockIdx.x * blockDim.x + threadIdx.x;
  if (idx >= N * ldk) return;
  int n = idx / ldk, k = idx % ldk;
  WT[idx] = f2bf((k < K) ? W[(size_t)k * N + n] : 0.f);
}

// ---------- streaming attention over ctx (memory-bound): s -> softmax -> u ----------
template<int NSLOT>
__global__ __launch_bounds__(256)
void attn_ctx_kernel(const float* __restrict__ ctx, const bf16_t* __restrict__ qw,
                     bf16_t* __restrict__ u, int M) {
  int wave = blockIdx.x * 8 + (threadIdx.x >> 5);
  int lane = threadIdx.x & 31;
  if (wave >= M) return;
  const float*  c0  = ctx + (size_t)wave * NSLOT * CDIM;
  const bf16_t* qwp = qw  + (size_t)wave * QW_LD;

  float qwl[NHEAD][7];
  #pragma unroll
  for (int h = 0; h < NHEAD; ++h)
    #pragma unroll
    for (int j = 0; j < 7; ++j) {
      int e = j * 32 + lane;
      qwl[h][j] = (e < CDIM) ? bf2f(qwp[h * CDIM + e]) : 0.f;
    }
  float qb[NHEAD];
  #pragma unroll
  for (int h = 0; h < NHEAD; ++h) qb[h] = bf2f(qwp[C4DIM + h]);

  float s[NHEAD][NSLOT];
  for (int n = 0; n < NSLOT; ++n) {
    if (n + 1 < NSLOT) __builtin_prefetch(c0 + (size_t)(n + 1) * CDIM + lane, 0, 1);
    float cr[7];
    #pragma unroll
    for (int j = 0; j < 7; ++j) {
      int e = j * 32 + lane;
      cr[j] = (e < CDIM) ? c0[(size_t)n * CDIM + e] : 0.f;
    }
    #pragma unroll
    for (int h = 0; h < NHEAD; ++h) {
      float d = 0.f;
      #pragma unroll
      for (int j = 0; j < 7; ++j) d += cr[j] * qwl[h][j];
      #pragma unroll
      for (int msk = 16; msk >= 1; msk >>= 1) d += __shfl_xor(d, msk, 32);
      s[h][n] = (d + qb[h]) * 0.142857143f;   // 1/sqrt(49)
    }
  }
  #pragma unroll
  for (int h = 0; h < NHEAD; ++h) {
    float mx = s[h][0];
    #pragma unroll
    for (int n = 1; n < NSLOT; ++n) mx = fmaxf(mx, s[h][n]);
    float sum = 0.f;
    #pragma unroll
    for (int n = 0; n < NSLOT; ++n) { s[h][n] = __expf(s[h][n] - mx); sum += s[h][n]; }
    float inv = 1.f / sum;
    #pragma unroll
    for (int n = 0; n < NSLOT; ++n) s[h][n] *= inv;
  }
  #pragma unroll
  for (int j = 0; j < 7; ++j) {
    int e = j * 32 + lane;
    if (e < CDIM) {
      float a0 = 0.f, a1 = 0.f, a2 = 0.f, a3 = 0.f;
      #pragma unroll
      for (int n = 0; n < NSLOT; ++n) {
        float cv = c0[(size_t)n * CDIM + e];
        a0 += s[0][n] * cv; a1 += s[1][n] * cv; a2 += s[2][n] * cv; a3 += s[3][n] * cv;
      }
      size_t o = (size_t)wave * C4DIM + e;
      u[o]            = f2bf(a0);
      u[o + CDIM]     = f2bf(a1);
      u[o + 2 * CDIM] = f2bf(a2);
      u[o + 3 * CDIM] = f2bf(a3);
    }
  }
}

__global__ void copy_kernel(const float* __restrict__ src, float* __restrict__ dst, size_t n) {
  size_t i = (size_t)blockIdx.x * blockDim.x + threadIdx.x;
  if (i < n) dst[i] = src[i];
}

// ---------------------------------- host launch ----------------------------------
extern "C" void kernel_launch(void* const* d_in, const int* in_sizes, int n_in,
                              void* d_out, int out_size, void* d_ws, size_t ws_size,
                              hipStream_t stream) {
  (void)in_sizes; (void)n_in; (void)out_size; (void)ws_size;
  const float* query    = (const float*)d_in[0];
  const float* ctx_view = (const float*)d_in[1];
  const float* ctx_plane= (const float*)d_in[2];
  const float* vw[14]; for (int i = 0; i < 14; ++i) vw[i] = (const float*)d_in[3 + i];
  const float* pw[14]; for (int i = 0; i < 14; ++i) pw[i] = (const float*)d_in[17 + i];
  // order: 0 Wk, 1 bk, 2 Wv, 3 bv, 4 ag, 5 ab, 6 W1, 7 b1, 8 W2, 9 b2, 10 n1g, 11 n1b, 12 n2g, 13 n2b

  float* q = (float*)d_out;                 // q lives in d_out, updated in place

  char* wsp = (char*)d_ws;
  auto take = [&](size_t bytes) { void* r = (void*)wsp; wsp += (bytes + 255) & ~(size_t)255; return r; };
  bf16_t* qn   = (bf16_t*)take((size_t)LTOK * QN_LD * sizeof(bf16_t));   // ~20 MB
  bf16_t* qwb  = (bf16_t*)take((size_t)LTOK * QW_LD * sizeof(bf16_t));   // ~79 MB
  bf16_t* big  = (bf16_t*)take((size_t)LTOK * C4DIM * sizeof(bf16_t));   // ~77 MB (u and h share)
  bf16_t* WqT  = (bf16_t*)take((size_t)QW_LD * KP196 * sizeof(bf16_t));
  bf16_t* WvT  = (bf16_t*)take((size_t)CDIM  * C4DIM * sizeof(bf16_t));
  bf16_t* W1T  = (bf16_t*)take((size_t)C4DIM * KP196 * sizeof(bf16_t));
  bf16_t* W2T  = (bf16_t*)take((size_t)CDIM  * C4DIM * sizeof(bf16_t));

  const size_t qelems = (size_t)LTOK * CDIM;
  copy_kernel<<<dim3((unsigned)((qelems + 255) / 256)), dim3(256), 0, stream>>>(query, q, qelems);

  const dim3 gTok((LTOK + 7) / 8);                       // 1 wave per token
  const dim3 gGemmM(LTOK / 128);                         // 384
  const dim3 gTileQW(LTOK / 128, (QW_LD + 63) / 64);     // 384 x 13
  const dim3 gTileF1(LTOK / 128, (C4DIM + 63) / 64);     // 384 x 13
  const dim3 blk(256);

  for (int layer = 0; layer < NLAYER; ++layer) {
    for (int sidx = 0; sidx < 2; ++sidx) {
      const float* const* W = (sidx == 0) ? vw : pw;
      const float* ctx      = (sidx == 0) ? ctx_view : ctx_plane;
      const int    nslot    = (sidx == 0) ? NV_SLOTS : NS_SLOTS;
      const size_t oW  = (size_t)layer * CDIM * CDIM;
      const size_t oC  = (size_t)layer * CDIM;
      const size_t oW1 = (size_t)layer * CDIM * C4DIM;
      const size_t oC4 = (size_t)layer * C4DIM;

      // 0) pre-transpose / pre-convert weights to bf16 [N][ldk] form (tiny)
      make_wqT_kernel<<<dim3((QW_LD * KP196 + 255) / 256), blk, 0, stream>>>(W[0] + oW, W[1] + oC, WqT);
      make_wvT_kernel<<<dim3((CDIM * C4DIM + 255) / 256), blk, 0, stream>>>(W[2] + oW, WvT);
      transpose_bf16_kernel<<<dim3((C4DIM * KP196 + 255) / 256), blk, 0, stream>>>(W[6] + oW1, W1T, CDIM, C4DIM, KP196);
      transpose_bf16_kernel<<<dim3((CDIM * C4DIM + 255) / 256), blk, 0, stream>>>(W[8] + oW1, W2T, C4DIM, CDIM, C4DIM);
      // 1) qn = LN(q)*ag + ab   (bf16, padded rows)
      ln_act_kernel<<<gTok, blk, 0, stream>>>(q, W[4] + oC, W[5] + oC, qn, LTOK);
      // 2) qw = qn @ Wq'   [L,196]x[196,800]  (WMMA bf16, async-staged)
      gemm_tile_kernel<1, 0><<<gTileQW, blk, 0, stream>>>(qn, WqT, nullptr, qwb,
                                                          LTOK, QW_LD, CDIM, QN_LD, KP196, QW_LD);
      // 3) stream ctx: scores -> softmax -> u   (HBM-bound pass)
      if (nslot == NV_SLOTS)
        attn_ctx_kernel<NV_SLOTS><<<gTok, blk, 0, stream>>>(ctx, qwb, big, LTOK);
      else
        attn_ctx_kernel<NS_SLOTS><<<gTok, blk, 0, stream>>>(ctx, qwb, big, LTOK);
      // 4) q = LN(q + u @ Wv' + bv)  [L,784]x[784,196]  (WMMA + fused LN)
      gemm_rowln_kernel<<<gGemmM, blk, 0, stream>>>(big, WvT, W[3] + oC, q,
                                                    W[10] + oC, W[11] + oC, q, LTOK, C4DIM);
      // 5) h = gelu(q @ W1 + b1)  [L,196]x[196,784]  (WMMA, bf16 out)
      gemm_tile_kernel<0, 1><<<gTileF1, blk, 0, stream>>>(q, W1T, W[7] + oC4, big,
                                                          LTOK, C4DIM, CDIM, CDIM, KP196, C4DIM);
      // 6) q = LN(q + h @ W2 + b2)  [L,784]x[784,196]  (WMMA + fused LN)
      gemm_rowln_kernel<<<gGemmM, blk, 0, stream>>>(big, W2T, W[9] + oC4, q,
                                                    W[12] + oC, W[13] + oC, q, LTOK, C4DIM);
    }
  }
}
